// GraphScoreNet_67602785239520
// MI455X (gfx1250) — matrix operand order repaired and checked by reading, not compiled, not run on hardware
//
#include <hip/hip_runtime.h>
#include <hip/hip_bf16.h>

// ---------------- problem constants (match reference) ----------------
static const int BB     = 8;
static const int NN     = 4096;
static const int KKNN   = 20;
static const int EEDG   = NN * KKNN;        // 81920 edges
static const int LL     = 128;              // latent
static const int DCOND  = 34;               // 32 timestep emb + 2 cond
static const int EINW   = 3 * LL + DCOND;   // 418 (valid cols)
static const int NINW   = 2 * LL + DCOND;   // 290 (valid cols)
static const int EINP   = 448;              // padded stride
static const int NINP   = 320;              // padded stride
static const int DP     = 32;               // padded stride for 3-wide inputs
static const int NSTEPS = 4;

typedef __attribute__((ext_vector_type(16))) _Float16     v16h;
typedef __attribute__((ext_vector_type(8)))  _Float16     v8h;
typedef __attribute__((ext_vector_type(8)))  float        v8f;
typedef __attribute__((ext_vector_type(4)))  unsigned int v4u;
typedef __attribute__((ext_vector_type(8)))  int          v8i;
typedef __attribute__((ext_vector_type(4)))  int          v4i;

__device__ __forceinline__ float gelu_tanh(float x) {
    // jax.nn.gelu default (approximate=True)
    float x3 = x * x * x;
    return 0.5f * x * (1.0f + tanhf(0.7978845608028654f * (x + 0.044715f * x3)));
}

// =====================================================================
// WMMA GEMM with TDM-fed, double-buffered LDS tiles.
//   Y = act(X[f16, M x Kpad, zero-padded] @ W + bias)
// WT: pre-transposed+padded f16 [16*NT][Kpad] (WT[n][k] = W[k][n]).
// Block = 256 threads (8 waves); block tile = 128 rows x 16*NT cols.
// Per K-chunk (32), wave 0 issues two tensor_load_to_lds DMAs (A and B
// tiles) into the *next* LDS buffer while all waves run WMMA on the
// current one. TDM pad feature reproduces the stride-40 LDS layout
// (16B pad after every 64B row) for conflict-free ds_load_b128.
// =====================================================================
static const int LDSP = 40; // 32 + 8 f16 pad

// Build D# and issue TENSOR_LOAD_TO_LDS for a 2D tile:
//   rows x 32 f16 elements, global row stride = Kpad elements,
//   LDS row stride = 80B (64B data + 16B pad via TDM padding).
__device__ __forceinline__ void tdm_tile_load(const _Float16* gsrc, unsigned int ldsAddr,
                                              int rows, int Kpad)
{
    unsigned long long ga = (unsigned long long)(const void*)gsrc;
    v4u g0;
    g0[0] = 1u;                                               // count=1 (valid descriptor)
    g0[1] = ldsAddr;                                          // lds_addr (bytes)
    g0[2] = (unsigned int)ga;                                 // global_addr[31:0]
    g0[3] = (unsigned int)((ga >> 32) & 0x01FFFFFFu)          // global_addr[56:32]
          | (2u << 30);                                       // type=2 ("image")
    v8i g1;
    g1[0] = (1 << 16)                                         // data_size = 2 bytes
          | (1 << 20)                                         // pad_enable
          | (3 << 22)                                         // pad_interval: 16 DWORDs (64B)
          | (3 << 25);                                        // pad_amount:   4 DWORDs (16B)
    g1[1] = (Kpad & 0xFFFF) << 16;                            // tensor_dim0[15:0]
    g1[2] = (rows & 0xFFFF) << 16;                            // tensor_dim0 hi | tensor_dim1[15:0]
    g1[3] = (32 << 16);                                       // tensor_dim1 hi | tile_dim0 = 32
    g1[4] = rows & 0xFFFF;                                    // tile_dim1 = rows, tile_dim2 = 0
    g1[5] = Kpad;                                             // tensor_dim0_stride (lo32)
    g1[6] = 0;
    g1[7] = 0;
    v4i g2 = (v4i)0, g3 = (v4i)0;                             // unused (<=2D tensor)
    v8i g4 = (v8i)0;                                          // unused extra group (6-arg form)
    __builtin_amdgcn_tensor_load_to_lds(g0, g1, g2, g3, g4, 0);
}

template <int NT, bool GELU_ACT, bool WF16, bool WF32>
__global__ __launch_bounds__(256)
void gemm_mlp_kernel(const _Float16* __restrict__ X, int M, int Kpad,
                     const _Float16* __restrict__ WT,
                     const float* __restrict__ bias, int Nout,
                     _Float16* __restrict__ Y16, float* __restrict__ Y32)
{
    __shared__ __align__(16) _Float16 As[2][128 * LDSP];
    __shared__ __align__(16) _Float16 Bs[2][16 * NT * LDSP];

    const int tid  = threadIdx.x;
    const int lane = tid & 31;
    const int wv   = tid >> 5;
    const int blockRow = blockIdx.x * 128;

    const _Float16* Xrow = X + (size_t)blockRow * Kpad;
    const unsigned int ldsA0 = (unsigned int)(unsigned long long)(uintptr_t)&As[0][0];
    const unsigned int ldsA1 = (unsigned int)(unsigned long long)(uintptr_t)&As[1][0];
    const unsigned int ldsB0 = (unsigned int)(unsigned long long)(uintptr_t)&Bs[0][0];
    const unsigned int ldsB1 = (unsigned int)(unsigned long long)(uintptr_t)&Bs[1][0];

    v8f acc[NT];
#pragma unroll
    for (int t = 0; t < NT; ++t) acc[t] = (v8f)0.0f;

    const int rowA  = wv * 16 + (lane & 15);
    const int aHalf = (lane >> 4) * 8;   // A frag: lanes 0-15 -> K 0-7 & 16-23; 16-31 -> K 8-15 & 24-31
    const int bHalf = (lane >> 4) * 8;   // B frag: lanes 0-15 -> K 0-7 & 16-23; 16-31 -> K 8-15 & 24-31

    // ---- prologue: DMA first K-chunk into buffer 0 ----
    if (wv == 0) {
        tdm_tile_load(Xrow, ldsA0, 128, Kpad);
        tdm_tile_load(WT,   ldsB0, 16 * NT, Kpad);
    }
    __builtin_amdgcn_s_wait_tensorcnt(0);
    __syncthreads();

    int cur = 0;
    for (int kc = 0; kc < Kpad; kc += 32) {
        const int nxt = cur ^ 1;
        // ---- async DMA of next K-chunk into the other buffer ----
        if (kc + 32 < Kpad && wv == 0) {
            tdm_tile_load(Xrow + kc + 32, nxt ? ldsA1 : ldsA0, 128, Kpad);
            tdm_tile_load(WT   + kc + 32, nxt ? ldsB1 : ldsB0, 16 * NT, Kpad);
        }

        // ---- compute on current buffer ----
        const _Float16* ap = &As[cur][rowA * LDSP + aHalf];
        v8h aLo = *(const v8h*)ap;
        v8h aHi = *(const v8h*)(ap + 16);
        v16h a;
#pragma unroll
        for (int i = 0; i < 8; ++i) { a[i] = aLo[i]; a[i + 8] = aHi[i]; }

#pragma unroll
        for (int t = 0; t < NT; ++t) {
            const _Float16* bp = &Bs[cur][(t * 16 + (lane & 15)) * LDSP + bHalf];
            v8h bLo = *(const v8h*)bp;
            v8h bHi = *(const v8h*)(bp + 16);
            v16h bf;
#pragma unroll
            for (int i = 0; i < 8; ++i) { bf[i] = bLo[i]; bf[i + 8] = bHi[i]; }
            acc[t] = __builtin_amdgcn_wmma_f32_16x16x32_f16(
                false, a, false, bf, (short)0, acc[t], false, false);
        }

        // DMA for next buffer must land before anyone reads it
        __builtin_amdgcn_s_wait_tensorcnt(0);
        __syncthreads();
        cur = nxt;
    }

    // ---- epilogue: bias (+GELU), store f16 and/or f32 ----
    const int mBase  = blockRow + wv * 16 + ((lane >> 4) << 3);
    const int nLocal = lane & 15;
#pragma unroll
    for (int t = 0; t < NT; ++t) {
        int n = t * 16 + nLocal;
        float bn = (n < Nout) ? bias[n] : 0.0f;
#pragma unroll
        for (int r = 0; r < 8; ++r) {
            int m = mBase + r;
            float v = acc[t][r] + bn;
            if (GELU_ACT) v = gelu_tanh(v);
            if (n < Nout) {
                size_t o = (size_t)m * Nout + n;
                if (WF16) Y16[o] = (_Float16)v;
                if (WF32) Y32[o] = v;
            }
        }
    }
}

// ---------------------------------------------------------------------
// weight prep: WT[n][k] = (f16)W[k][n], zero pad to [NoutPad][Kpad]
// ---------------------------------------------------------------------
__global__ void prep_wt_kernel(const float* __restrict__ W, _Float16* __restrict__ WT,
                               int Kdim, int Nout, int Kpad, int NoutPad)
{
    int gid = blockIdx.x * 256 + threadIdx.x;
    if (gid >= NoutPad * Kpad) return;
    int n = gid / Kpad, k = gid % Kpad;
    _Float16 v = (_Float16)0.0f;
    if (n < Nout && k < Kdim) v = (_Float16)W[(size_t)k * Nout + n];
    WT[gid] = v;
}

// ---------------------------------------------------------------------
// cond MLP (tiny, B=8): timestep embedding -> [34,136,136,34] MLP -> f16
// ---------------------------------------------------------------------
__global__ __launch_bounds__(256)
void cond_kernel(const float* __restrict__ t, const float* __restrict__ cnd,
                 const float* __restrict__ w0, const float* __restrict__ b0,
                 const float* __restrict__ w1, const float* __restrict__ b1,
                 const float* __restrict__ w2, const float* __restrict__ b2,
                 _Float16* __restrict__ cond16)
{
    __shared__ float xs[BB * 136];
    __shared__ float ys[BB * 136];
    const int tid = threadIdx.x;
    const float nlog = -logf(10000.0f);

    for (int i = tid; i < BB * DCOND; i += 256) {
        int b = i / DCOND, j = i % DCOND;
        float v;
        if (j < 16)      v = sinf(t[b] * expf(nlog * (float)j / 15.0f));
        else if (j < 32) v = cosf(t[b] * expf(nlog * (float)(j - 16) / 15.0f));
        else             v = cnd[b * 2 + (j - 32)];
        xs[b * 136 + j] = v;
    }
    __syncthreads();
    for (int i = tid; i < BB * 136; i += 256) {
        int b = i / 136, o = i % 136;
        float s = b0[o];
        for (int k = 0; k < DCOND; ++k) s += xs[b * 136 + k] * w0[k * 136 + o];
        ys[i] = gelu_tanh(s);
    }
    __syncthreads();
    for (int i = tid; i < BB * 136; i += 256) {
        int b = i / 136, o = i % 136;
        float s = b1[o];
        for (int k = 0; k < 136; ++k) s += ys[b * 136 + k] * w1[k * 136 + o];
        xs[i] = gelu_tanh(s);
    }
    __syncthreads();
    for (int i = tid; i < BB * DCOND; i += 256) {
        int b = i / DCOND, o = i % DCOND;
        float s = b2[o];
        for (int k = 0; k < 136; ++k) s += xs[b * 136 + k] * w2[k * 34 + o];
        cond16[i] = (_Float16)s;
    }
}

// ---------------------------------------------------------------------
// kNN: one thread per node; LDS-tiled candidates; register top-20
// ---------------------------------------------------------------------
static const int KTILE = 512;
__global__ __launch_bounds__(128)
void knn_kernel(const float* __restrict__ z, const float* __restrict__ xmean,
                const float* __restrict__ xstd, int* __restrict__ idxOut)
{
    __shared__ float xs[KTILE * 3];
    const int i = blockIdx.x * 128 + threadIdx.x;
    const float sx = xstd[0], sy = xstd[1], sz = xstd[2];
    const float mx = xmean[0], my = xmean[1], mz = xmean[2];
    const float xi = z[i * 3 + 0] * sx + mx;
    const float yi = z[i * 3 + 1] * sy + my;
    const float zi = z[i * 3 + 2] * sz + mz;

    float bd[KKNN]; int bi[KKNN];
#pragma unroll
    for (int q = 0; q < KKNN; ++q) { bd[q] = 3.0e38f; bi[q] = 0; }

    for (int base = 0; base < NN; base += KTILE) {
        for (int j = threadIdx.x; j < KTILE * 3; j += 128)
            xs[j] = z[base * 3 + j];
        __syncthreads();
        for (int j = 0; j < KTILE; ++j) {
            float dx = xi - (xs[j * 3 + 0] * sx + mx);
            float dy = yi - (xs[j * 3 + 1] * sy + my);
            float dz = zi - (xs[j * 3 + 2] * sz + mz);
            float d = dx * dx + dy * dy + dz * dz;
            if (d < bd[KKNN - 1]) {
                bd[KKNN - 1] = d; bi[KKNN - 1] = base + j;
#pragma unroll
                for (int q = KKNN - 1; q > 0; --q) {
                    float dq = bd[q], dp = bd[q - 1];
                    int   iq = bi[q], ip = bi[q - 1];
                    bool sw = dq < dp;
                    bd[q - 1] = sw ? dq : dp; bi[q - 1] = sw ? iq : ip;
                    bd[q]     = sw ? dp : dq; bi[q]     = sw ? ip : iq;
                }
            }
        }
        __syncthreads();
    }
#pragma unroll
    for (int q = 0; q < KKNN; ++q) idxOut[i * KKNN + q] = bi[q];
}

// displacement edge features, padded to DP cols: dr = z[src]-z[tgt], 0 pad
__global__ void edge_dr_kernel(const float* __restrict__ z, const int* __restrict__ idx,
                               _Float16* __restrict__ dr)
{
    int gid = blockIdx.x * 256 + threadIdx.x;
    if (gid >= EEDG * DP) return;
    int e = gid / DP, j = gid % DP;
    _Float16 v = (_Float16)0.0f;
    if (j < 3) {
        int i = e / KKNN, k = e % KKNN;
        int s = idx[i * KKNN + 0];
        int t = idx[i * KKNN + k];
        v = (_Float16)(z[s * 3 + j] - z[t * 3 + j]);
    }
    dr[gid] = v;
}

// node positions, padded to DP cols
__global__ void node_z_kernel(const float* __restrict__ z, _Float16* __restrict__ zn)
{
    int gid = blockIdx.x * 256 + threadIdx.x;
    if (gid >= NN * DP) return;
    int i = gid / DP, j = gid % DP;
    zn[gid] = (j < 3) ? (_Float16)z[i * 3 + j] : (_Float16)0.0f;
}

// e_in = [e, h[src], h[tgt], cond, 0-pad]  (f16, stride EINP)
__global__ void gather_edge_in_kernel(const _Float16* __restrict__ e16,
                                      const _Float16* __restrict__ h16,
                                      const int* __restrict__ idx,
                                      const _Float16* __restrict__ cond16,
                                      _Float16* __restrict__ ein)
{
    size_t gid = (size_t)blockIdx.x * 256 + threadIdx.x;
    if (gid >= (size_t)EEDG * EINP) return;
    int e = (int)(gid / EINP), j = (int)(gid % EINP);
    int i = e / KKNN, k = e % KKNN;
    _Float16 v = (_Float16)0.0f;
    if (j < 128)      v = e16[(size_t)e * 128 + j];
    else if (j < 256) v = h16[(size_t)idx[i * KKNN + 0] * 128 + (j - 128)];
    else if (j < 384) v = h16[(size_t)idx[i * KKNN + k] * 128 + (j - 256)];
    else if (j < EINW) v = cond16[j - 384];
    ein[gid] = v;
}

// n_in = [h, agg, cond, 0-pad]  (f16, stride NINP)
__global__ void gather_node_in_kernel(const _Float16* __restrict__ h16,
                                      const float* __restrict__ agg32,
                                      const _Float16* __restrict__ cond16,
                                      _Float16* __restrict__ nin)
{
    size_t gid = (size_t)blockIdx.x * 256 + threadIdx.x;
    if (gid >= (size_t)NN * NINP) return;
    int i = (int)(gid / NINP), j = (int)(gid % NINP);
    _Float16 v = (_Float16)0.0f;
    if (j < 128)      v = h16[(size_t)i * 128 + j];
    else if (j < 256) v = (_Float16)agg32[(size_t)i * 128 + (j - 128)];
    else if (j < NINW) v = cond16[j - 256];
    nin[gid] = v;
}

__global__ void zero_f32_kernel(float* __restrict__ p, size_t n)
{
    size_t gid = (size_t)blockIdx.x * 256 + threadIdx.x;
    if (gid < n) p[gid] = 0.0f;
}

// segment_sum over targets: agg[tgt] += e_new  (f32 atomics)
__global__ void scatter_agg_kernel(const float* __restrict__ enew, const int* __restrict__ idx,
                                   float* __restrict__ agg32)
{
    size_t gid = (size_t)blockIdx.x * 256 + threadIdx.x;
    if (gid >= (size_t)EEDG * 128) return;
    int e = (int)(gid >> 7), j = (int)(gid & 127);
    int i = e / KKNN, k = e % KKNN;
    int t = idx[i * KKNN + k];
    atomicAdd(&agg32[(size_t)t * 128 + j], enew[gid]);
}

// x = LayerNorm(x + xnew) * g + b   (updates f32 master + f16 mirror)
__global__ __launch_bounds__(256)
void skip_ln_kernel(float* __restrict__ x32, _Float16* __restrict__ x16,
                    const float* __restrict__ xnew, const float* __restrict__ g,
                    const float* __restrict__ bta, int M)
{
    int row  = blockIdx.x * 8 + (threadIdx.x >> 5);
    int lane = threadIdx.x & 31;
    if (row >= M) return;
    size_t rb = (size_t)row * LL;
    float v[4];
    float s = 0.0f;
#pragma unroll
    for (int c = 0; c < 4; ++c) {
        int j = lane + c * 32;
        v[c] = x32[rb + j] + xnew[rb + j];
        s += v[c];
    }
#pragma unroll
    for (int o = 16; o > 0; o >>= 1) s += __shfl_xor(s, o, 32);
    float mean = s * (1.0f / LL);
    float s2 = 0.0f;
#pragma unroll
    for (int c = 0; c < 4; ++c) { float d = v[c] - mean; s2 += d * d; }
#pragma unroll
    for (int o = 16; o > 0; o >>= 1) s2 += __shfl_xor(s2, o, 32);
    float inv = rsqrtf(s2 * (1.0f / LL) + 1e-6f);
#pragma unroll
    for (int c = 0; c < 4; ++c) {
        int j = lane + c * 32;
        float y = (v[c] - mean) * inv * g[j] + bta[j];
        x32[rb + j] = y;
        x16[rb + j] = (_Float16)y;
    }
}

// eps = z - decoded
__global__ void eps_kernel(const float* __restrict__ z, const float* __restrict__ dec,
                           float* __restrict__ out)
{
    int gid = blockIdx.x * 256 + threadIdx.x;
    if (gid < NN * 3) out[gid] = z[gid] - dec[gid];
}

// =====================================================================
// host orchestration
// =====================================================================
extern "C" void kernel_launch(void* const* d_in, const int* in_sizes, int n_in,
                              void* d_out, int out_size, void* d_ws, size_t ws_size,
                              hipStream_t stream)
{
    (void)in_sizes; (void)n_in; (void)out_size; (void)ws_size;

    const float* z      = (const float*)d_in[0];
    const float* t_in   = (const float*)d_in[1];
    const float* cnd    = (const float*)d_in[2];
    /* d_in[3] = mask (all true) ignored */
    const float* x_mean = (const float*)d_in[4];
    const float* x_std  = (const float*)d_in[5];

    // ---- pytree leaf indexing (dict keys sorted, tuples (w,b) in order) ----
    int p = 6;
    int cw[3], cb[3];
    for (int l = 0; l < 3; ++l) { cw[l] = p++; cb[l] = p++; }          // cond_mlp
    int dw[4], db[4];
    for (int l = 0; l < 4; ++l) { dw[l] = p++; db[l] = p++; }          // decoder
    int ew[4], eb[4];
    for (int l = 0; l < 4; ++l) { ew[l] = p++; eb[l] = p++; }          // edge_embed
    int nw[4], nb[4];
    for (int l = 0; l < 4; ++l) { nw[l] = p++; nb[l] = p++; }          // node_embed
    int sew[NSTEPS][4], seb[NSTEPS][4], lneg[NSTEPS], lneb[NSTEPS],
        lnng[NSTEPS], lnnb[NSTEPS], snw[NSTEPS][4], snb[NSTEPS][4];
    for (int s = 0; s < NSTEPS; ++s) {                                  // steps
        for (int l = 0; l < 4; ++l) { sew[s][l] = p++; seb[s][l] = p++; } // edge_mlp
        lneg[s] = p++; lneb[s] = p++;                                     // ln_e (g,b)
        lnng[s] = p++; lnnb[s] = p++;                                     // ln_n (g,b)
        for (int l = 0; l < 4; ++l) { snw[s][l] = p++; snb[s][l] = p++; } // node_mlp
    }
    auto F = [&](int i) { return (const float*)d_in[i]; };

    // ---- workspace bump allocator ----
    char* base = (char*)d_ws;
    size_t off = 0;
    auto alloc = [&](size_t bytes) -> char* {
        off = (off + 255) & ~(size_t)255;
        char* r = base + off; off += bytes; return r;
    };
    auto allocWT = [&](int NoutPad, int Kpad) {
        return (_Float16*)alloc((size_t)NoutPad * Kpad * 2);
    };

    _Float16* cond16 = (_Float16*)alloc((size_t)BB * DCOND * 2);

    _Float16 *wtNE[4], *wtEE[4], *wtDEC[4], *wtSE[NSTEPS][4], *wtSN[NSTEPS][4];
    wtNE[0] = allocWT(128, DP);
    for (int l = 1; l < 4; ++l) wtNE[l] = allocWT(128, 128);
    wtEE[0] = allocWT(128, DP);
    for (int l = 1; l < 4; ++l) wtEE[l] = allocWT(128, 128);
    for (int l = 0; l < 3; ++l) wtDEC[l] = allocWT(128, 128);
    wtDEC[3] = allocWT(16, 128);
    for (int s = 0; s < NSTEPS; ++s) {
        wtSE[s][0] = allocWT(128, EINP);
        for (int l = 1; l < 4; ++l) wtSE[s][l] = allocWT(128, 128);
        wtSN[s][0] = allocWT(128, NINP);
        for (int l = 1; l < 4; ++l) wtSN[s][l] = allocWT(128, 128);
    }

    int*      idxb   = (int*)alloc((size_t)NN * KKNN * 4);
    _Float16* zn16   = (_Float16*)alloc((size_t)NN * DP * 2);
    _Float16* dr16   = (_Float16*)alloc((size_t)EEDG * DP * 2);
    float*    h32    = (float*)alloc((size_t)NN * LL * 4);
    _Float16* h16    = (_Float16*)alloc((size_t)NN * LL * 2);
    float*    e32    = (float*)alloc((size_t)EEDG * LL * 4);
    _Float16* e16    = (_Float16*)alloc((size_t)EEDG * LL * 2);
    _Float16* ein16  = (_Float16*)alloc((size_t)EEDG * EINP * 2);
    _Float16* nin16  = (_Float16*)alloc((size_t)NN * NINP * 2);
    _Float16* tA     = (_Float16*)alloc((size_t)EEDG * LL * 2);
    _Float16* tB     = (_Float16*)alloc((size_t)EEDG * LL * 2);
    float*    enew32 = (float*)alloc((size_t)EEDG * LL * 4);
    float*    hnew32 = (float*)alloc((size_t)NN * LL * 4);
    float*    agg32  = (float*)alloc((size_t)NN * LL * 4);
    float*    dec32  = (float*)alloc((size_t)NN * 3 * 4);

    // ---- weight prep (fp32 -> transposed padded f16) ----
    auto prep = [&](int wi, _Float16* WT, int Kd, int No, int Kp, int Np) {
        int n = Np * Kp;
        prep_wt_kernel<<<(n + 255) / 256, 256, 0, stream>>>(F(wi), WT, Kd, No, Kp, Np);
    };
    prep(nw[0], wtNE[0], 3, 128, DP, 128);
    for (int l = 1; l < 4; ++l) prep(nw[l], wtNE[l], 128, 128, 128, 128);
    prep(ew[0], wtEE[0], 3, 128, DP, 128);
    for (int l = 1; l < 4; ++l) prep(ew[l], wtEE[l], 128, 128, 128, 128);
    for (int l = 0; l < 3; ++l) prep(dw[l], wtDEC[l], 128, 128, 128, 128);
    prep(dw[3], wtDEC[3], 128, 3, 128, 16);
    for (int s = 0; s < NSTEPS; ++s) {
        prep(sew[s][0], wtSE[s][0], EINW, 128, EINP, 128);
        for (int l = 1; l < 4; ++l) prep(sew[s][l], wtSE[s][l], 128, 128, 128, 128);
        prep(snw[s][0], wtSN[s][0], NINW, 128, NINP, 128);
        for (int l = 1; l < 4; ++l) prep(snw[s][l], wtSN[s][l], 128, 128, 128, 128);
    }

    // ---- cond MLP (once, all batches) ----
    cond_kernel<<<1, 256, 0, stream>>>(t_in, cnd, F(cw[0]), F(cb[0]),
                                       F(cw[1]), F(cb[1]), F(cw[2]), F(cb[2]), cond16);

    // ---- GEMM dispatcher (X stride == Kpad, zero-padded) ----
    auto gemm = [&](const _Float16* X, int M, int Kp, const _Float16* WT,
                    const float* bias, int Nout, int NoutPad, bool gelu,
                    _Float16* Y16, float* Y32) {
        dim3 g((M + 127) / 128), blk(256);
        if (NoutPad == 128) {
            if (gelu)
                gemm_mlp_kernel<8, true, true, false><<<g, blk, 0, stream>>>(
                    X, M, Kp, WT, bias, Nout, Y16, nullptr);
            else if (Y16 && Y32)
                gemm_mlp_kernel<8, false, true, true><<<g, blk, 0, stream>>>(
                    X, M, Kp, WT, bias, Nout, Y16, Y32);
            else
                gemm_mlp_kernel<8, false, false, true><<<g, blk, 0, stream>>>(
                    X, M, Kp, WT, bias, Nout, nullptr, Y32);
        } else {
            gemm_mlp_kernel<1, false, false, true><<<g, blk, 0, stream>>>(
                X, M, Kp, WT, bias, Nout, nullptr, Y32);
        }
    };

    // ---- per-batch pipeline ----
    for (int b = 0; b < BB; ++b) {
        const float* zb = z + (size_t)b * NN * 3;
        const _Float16* cb16 = cond16 + b * DCOND;
        float* outb = (float*)d_out + (size_t)b * NN * 3;

        knn_kernel<<<NN / 128, 128, 0, stream>>>(zb, x_mean, x_std, idxb);
        edge_dr_kernel<<<(EEDG * DP + 255) / 256, 256, 0, stream>>>(zb, idxb, dr16);
        node_z_kernel<<<(NN * DP + 255) / 256, 256, 0, stream>>>(zb, zn16);

        // node embed: [3 ->128 ->128 ->128 ->128], GELU between, last linear
        gemm(zn16, NN, DP,  wtNE[0], F(nb[0]), 128, 128, true,  tA, nullptr);
        gemm(tA,  NN, 128, wtNE[1], F(nb[1]), 128, 128, true,  tB, nullptr);
        gemm(tB,  NN, 128, wtNE[2], F(nb[2]), 128, 128, true,  tA, nullptr);
        gemm(tA,  NN, 128, wtNE[3], F(nb[3]), 128, 128, false, h16, h32);

        // edge embed on dr
        gemm(dr16, EEDG, DP,  wtEE[0], F(eb[0]), 128, 128, true,  tA, nullptr);
        gemm(tA,   EEDG, 128, wtEE[1], F(eb[1]), 128, 128, true,  tB, nullptr);
        gemm(tB,   EEDG, 128, wtEE[2], F(eb[2]), 128, 128, true,  tA, nullptr);
        gemm(tA,   EEDG, 128, wtEE[3], F(eb[3]), 128, 128, false, e16, e32);

        for (int s = 0; s < NSTEPS; ++s) {
            // edge MLP
            size_t etot = (size_t)EEDG * EINP;
            gather_edge_in_kernel<<<(unsigned)((etot + 255) / 256), 256, 0, stream>>>(
                e16, h16, idxb, cb16, ein16);
            gemm(ein16, EEDG, EINP, wtSE[s][0], F(seb[s][0]), 128, 128, true,  tA, nullptr);
            gemm(tA,    EEDG, 128,  wtSE[s][1], F(seb[s][1]), 128, 128, true,  tB, nullptr);
            gemm(tB,    EEDG, 128,  wtSE[s][2], F(seb[s][2]), 128, 128, true,  tA, nullptr);
            gemm(tA,    EEDG, 128,  wtSE[s][3], F(seb[s][3]), 128, 128, false, nullptr, enew32);

            // segment_sum over targets
            zero_f32_kernel<<<(NN * LL + 255) / 256, 256, 0, stream>>>(agg32, (size_t)NN * LL);
            scatter_agg_kernel<<<(unsigned)(((size_t)EEDG * 128 + 255) / 256), 256, 0, stream>>>(
                enew32, idxb, agg32);

            // node MLP
            size_t ntot = (size_t)NN * NINP;
            gather_node_in_kernel<<<(unsigned)((ntot + 255) / 256), 256, 0, stream>>>(
                h16, agg32, cb16, nin16);
            gemm(nin16, NN, NINP, wtSN[s][0], F(snb[s][0]), 128, 128, true,  tA, nullptr);
            gemm(tA,    NN, 128,  wtSN[s][1], F(snb[s][1]), 128, 128, true,  tB, nullptr);
            gemm(tB,    NN, 128,  wtSN[s][2], F(snb[s][2]), 128, 128, true,  tA, nullptr);
            gemm(tA,    NN, 128,  wtSN[s][3], F(snb[s][3]), 128, 128, false, nullptr, hnew32);

            // skip + layernorm (node, then edge)
            skip_ln_kernel<<<(NN + 7) / 8, 256, 0, stream>>>(h32, h16, hnew32,
                                                             F(lnng[s]), F(lnnb[s]), NN);
            skip_ln_kernel<<<(EEDG + 7) / 8, 256, 0, stream>>>(e32, e16, enew32,
                                                               F(lneg[s]), F(lneb[s]), EEDG);
        }

        // decoder: [128->128->128->128->3]
        gemm(h16, NN, 128, wtDEC[0], F(db[0]), 128, 128, true, tA, nullptr);
        gemm(tA,  NN, 128, wtDEC[1], F(db[1]), 128, 128, true, tB, nullptr);
        gemm(tB,  NN, 128, wtDEC[2], F(db[2]), 128, 128, true, tA, nullptr);
        gemm(tA,  NN, 128, wtDEC[3], F(db[3]), 3, 16, false, nullptr, dec32);

        eps_kernel<<<(NN * 3 + 255) / 256, 256, 0, stream>>>(zb, dec32, outb);
    }
}
